// MambaSurvivalBlock_68135361184267
// MI455X (gfx1250) — matrix operand order repaired and checked by reading, not compiled, not run on hardware
//
#include <hip/hip_runtime.h>
#include <hip/hip_bf16.h>

// ---------------------------------------------------------------------------
// Mamba block for gfx1250 (MI455X).
//   B=2, L=1024, d_model=1024, d_inner=2048, d_state=16, dt_rank=64, d_conv=4
// Pipeline:
//   1) in_proj GEMM (WMMA bf16-split, fused x/z split store)
//   2) depthwise causal conv1d + SiLU (VALU)
//   3) x_proj GEMM (WMMA)            -> x_dbl (dt_lo | B | C)
//   4) dt_proj GEMM (WMMA, fused softplus + bias)
//   5) selective scan (fused dA/dBx computation, D-skip, SiLU gate)
//   6) out_proj GEMM (WMMA) -> d_out
// GEMMs: fp32 via V_WMMA_F32_16X16X32_BF16 with 2-term bf16 split
// (hi*hi + hi*lo + lo*hi, fp32 accumulate). A-chunk staged in LDS per block;
// each wave computes two 16x16 N-tiles reusing its A fragments.
// ---------------------------------------------------------------------------

typedef __attribute__((ext_vector_type(16))) __bf16 v16bf;
typedef __attribute__((ext_vector_type(8)))  float  v8f;

#define D_MODEL 1024
#define D_STATE 16
#define D_CONV  4
#define D_INNER 2048
#define DT_RANK 64
#define BATCH   2
#define SEQLEN  1024
#define XDBL_W  (DT_RANK + 2 * D_STATE)   // 96

#define WMMA_BF16(A_, B_, C_) \
  __builtin_amdgcn_wmma_f32_16x16x32_bf16(false, (A_), false, (B_), (short)0, (C_), false, false)

// ---------------------------------------------------------------------------
// Generic wave-tile GEMM:  C[M,N] = A[M,K] * W[N,K]^T   (W row-major N x K)
// Block = 128 threads (4 waves) covering one M-tile and 8 N-tiles
// (2 N-tiles per wave). K consumed 32 at a time; A chunk staged in LDS.
// MODE 0: plain store. MODE 1: split store (cols < N/2 -> Cout, else Cout2).
// MODE 2: softplus(acc + bias[n]).
// ---------------------------------------------------------------------------
template <int MODE>
__global__ void gemm_wmma_kernel(const float* __restrict__ A, int lda,
                                 const float* __restrict__ W, int ldw,
                                 float* __restrict__ Cout, int ldc,
                                 float* __restrict__ Cout2,
                                 const float* __restrict__ bias,
                                 int M, int N, int K) {
  __shared__ float As[16][33];               // 16 x 32 A chunk, +1 pad vs banks

  const int lane = threadIdx.x & 31;
  const int wave = threadIdx.x >> 5;
  const int row0 = blockIdx.y * 16;
  const int col0 = (blockIdx.x * 8 + wave * 2) * 16;   // first of 2 N-tiles

  const int h  = lane >> 4;                  // half-wave select
  const int mr = lane & 15;                  // row (A) / col (B,C) within tile

  // Clamped W row indices for the two tiles: out-of-range tiles read row N-1
  // (harmless) and never store. All waves stay live for the LDS barriers.
  const int c0 = col0 + mr;
  const int c1 = col0 + 16 + mr;
  const int wr0 = (c0 < N) ? c0 : (N - 1);
  const int wr1 = (c1 < N) ? c1 : (N - 1);

  v8f acc0 = {}, acc1 = {};

  for (int k0 = 0; k0 < K; k0 += 32) {
    // --- cooperative A-chunk staging: 512 floats by 128 threads -----------
    {
      int i = threadIdx.x;
#pragma unroll
      for (int r = 0; r < 4; ++r, i += 128)
        As[i >> 5][i & 31] = A[(size_t)(row0 + (i >> 5)) * lda + k0 + (i & 31)];
    }
    __syncthreads();

    // --- A fragment: 16x32 bf16 layout (ISA 7.12.2), built from LDS -------
    // element i : K = 16h + i (contiguous, matches B fill).
    v16bf a_hi, a_lo;
#pragma unroll
    for (int i = 0; i < 16; ++i) {
      const int j = i >> 1, p = i & 1;
      const int kk = (h << 4) + i;
      const float v = As[mr][kk];
      const __bf16 hi = (__bf16)v;
      a_hi[i] = hi;
      a_lo[i] = (__bf16)(v - (float)hi);
    }

    // --- B fragments for two N-tiles: lane = N col, elements walk K -------
    // B[k][n] = W[col+n][k]; half-wave adds K+16 (contiguous k order, same as A).
    const float* w0 = W + (size_t)wr0 * ldw + k0 + (h << 4);
    const float* w1 = W + (size_t)wr1 * ldw + k0 + (h << 4);
    v16bf b0_hi, b0_lo, b1_hi, b1_lo;
#pragma unroll
    for (int i = 0; i < 16; ++i) {
      const float v0 = w0[i];
      const float v1 = w1[i];
      const __bf16 h0 = (__bf16)v0;
      const __bf16 h1 = (__bf16)v1;
      b0_hi[i] = h0; b0_lo[i] = (__bf16)(v0 - (float)h0);
      b1_hi[i] = h1; b1_lo[i] = (__bf16)(v1 - (float)h1);
    }

    if (k0 + 32 < K) {
      __builtin_prefetch(w0 + 32, 0, 3);
      __builtin_prefetch(w1 + 32, 0, 3);
    }

    // --- 6 WMMAs per chunk: fp32-accurate split product (drop lo*lo) ------
    acc0 = WMMA_BF16(a_hi, b0_hi, acc0);
    acc1 = WMMA_BF16(a_hi, b1_hi, acc1);
    acc0 = WMMA_BF16(a_hi, b0_lo, acc0);
    acc1 = WMMA_BF16(a_hi, b1_lo, acc1);
    acc0 = WMMA_BF16(a_lo, b0_hi, acc0);
    acc1 = WMMA_BF16(a_lo, b1_hi, acc1);

    __syncthreads();                         // protect As before next stage
  }

  // --- epilogue: C/D layout VGPR r -> M = r + 8h, N = lane&15 -------------
#pragma unroll
  for (int t = 0; t < 2; ++t) {
    const int tcol = col0 + t * 16;
    if (tcol >= N) continue;                 // wave-uniform skip
    const int n = tcol + mr;
    const v8f acc = t ? acc1 : acc0;
#pragma unroll
    for (int r = 0; r < 8; ++r) {
      const int m = row0 + r + (h << 3);
      const float v = acc[r];
      if (MODE == 0) {
        Cout[(size_t)m * ldc + n] = v;
      } else if (MODE == 1) {
        const int half = N >> 1;
        if (n < half) Cout [(size_t)m * ldc + n]          = v;
        else          Cout2[(size_t)m * ldc + (n - half)] = v;
      } else {  // MODE 2: softplus(acc + bias)
        const float x  = v + bias[n];
        const float sp = (x > 20.0f) ? x : __logf(1.0f + __expf(x));
        Cout[(size_t)m * ldc + n] = sp;
      }
    }
  }
}

// ---------------------------------------------------------------------------
// Depthwise causal conv1d (kernel 4) + bias + SiLU
// ---------------------------------------------------------------------------
__global__ void conv_silu_kernel(const float* __restrict__ x_in,
                                 const float* __restrict__ conv_w,
                                 const float* __restrict__ conv_b,
                                 float* __restrict__ x_conv) {
  const int idx = blockIdx.x * blockDim.x + threadIdx.x;
  const int total = BATCH * SEQLEN * D_INNER;
  if (idx >= total) return;
  const int c = idx % D_INNER;
  const int t = (idx / D_INNER) % SEQLEN;
  const int b = idx / (D_INNER * SEQLEN);

  float acc = conv_b[c];
#pragma unroll
  for (int k = 0; k < D_CONV; ++k) {
    const int ts = t + k - (D_CONV - 1);
    if (ts >= 0)
      acc += x_in[((size_t)(b * SEQLEN + ts)) * D_INNER + c] * conv_w[c * D_CONV + k];
  }
  x_conv[idx] = acc / (1.0f + __expf(-acc));   // SiLU
}

// ---------------------------------------------------------------------------
// Selective scan, fused: dA/dBx recomputed on the fly (saves 512MB of
// intermediates), D-skip and SiLU(z) gate applied inline.
// One thread per (batch, channel); 16-state kept in registers.
// ---------------------------------------------------------------------------
__global__ void scan_kernel(const float* __restrict__ dt,
                            const float* __restrict__ x_conv,
                            const float* __restrict__ x_dbl,
                            const float* __restrict__ zbuf,
                            const float* __restrict__ A_log,
                            const float* __restrict__ Dp,
                            float* __restrict__ y) {
  const int idx = blockIdx.x * blockDim.x + threadIdx.x;
  if (idx >= BATCH * D_INNER) return;
  const int c = idx % D_INNER;
  const int b = idx / D_INNER;

  float Arow[D_STATE];
#pragma unroll
  for (int s = 0; s < D_STATE; ++s) Arow[s] = -__expf(A_log[c * D_STATE + s]);
  const float Dc = Dp[c];

  float h[D_STATE];
#pragma unroll
  for (int s = 0; s < D_STATE; ++s) h[s] = 0.0f;

  for (int t = 0; t < SEQLEN; ++t) {
    const size_t row = (size_t)(b * SEQLEN + t);
    const float dtv = dt[row * D_INNER + c];
    const float xv  = x_conv[row * D_INNER + c];
    const float* bd = x_dbl + row * XDBL_W;   // [0:64)=dt_lo, [64:80)=B, [80:96)=C

    float yv = 0.0f;
#pragma unroll
    for (int s = 0; s < D_STATE; ++s) {
      const float dA  = __expf(dtv * Arow[s]);
      const float dBx = dtv * bd[DT_RANK + s] * xv;
      h[s] = dA * h[s] + dBx;
      yv += h[s] * bd[DT_RANK + D_STATE + s];
    }
    const float zv = zbuf[row * D_INNER + c];
    const float gate = zv / (1.0f + __expf(-zv));     // SiLU(z)
    y[row * D_INNER + c] = (yv + Dc * xv) * gate;
  }
}

// ---------------------------------------------------------------------------
extern "C" void kernel_launch(void* const* d_in, const int* in_sizes, int n_in,
                              void* d_out, int out_size, void* d_ws, size_t ws_size,
                              hipStream_t stream) {
  (void)in_sizes; (void)n_in; (void)out_size; (void)ws_size;

  const float* x      = (const float*)d_in[0];   // (B,L,1024)
  const float* W_in   = (const float*)d_in[1];   // (4096,1024)
  const float* conv_w = (const float*)d_in[2];   // (2048,4)
  const float* conv_b = (const float*)d_in[3];   // (2048,)
  const float* W_xp   = (const float*)d_in[4];   // (96,2048)
  const float* W_dt   = (const float*)d_in[5];   // (2048,64)
  const float* b_dt   = (const float*)d_in[6];   // (2048,)
  const float* A_log  = (const float*)d_in[7];   // (2048,16)
  const float* Dp     = (const float*)d_in[8];   // (2048,)
  const float* W_out  = (const float*)d_in[9];   // (1024,2048)
  float* out = (float*)d_out;                    // (B,L,1024)

  const int M = BATCH * SEQLEN;                  // 2048

  // Workspace layout (floats); ~85 MB total
  float* ws      = (float*)d_ws;
  const size_t NBL = (size_t)M * D_INNER;        // 4,194,304
  float* x_in    = ws;                           // 16 MB
  float* zbuf    = x_in + NBL;                   // 16 MB
  float* x_conv  = zbuf + NBL;                   // 16 MB
  float* x_dbl   = x_conv + NBL;                 // (M,96)  0.75 MB
  float* dtbuf   = x_dbl + (size_t)M * XDBL_W;   // 16 MB
  float* ybuf    = dtbuf + NBL;                  // 16 MB

  const dim3 blk(128);                           // 4 waves; 8 N-tiles / block

  // 1) in_proj: (2048 x 4096) = x(2048x1024) * W_in^T ; split -> x_in | z
  {
    const int N = 2 * D_INNER, K = D_MODEL;      // 256 N-tiles
    dim3 grid((N / 16 + 7) / 8, M / 16);
    gemm_wmma_kernel<1><<<grid, blk, 0, stream>>>(x, D_MODEL, W_in, D_MODEL,
                                                  x_in, D_INNER, zbuf, nullptr,
                                                  M, N, K);
  }

  // 2) depthwise causal conv + SiLU
  {
    const int total = BATCH * SEQLEN * D_INNER;
    conv_silu_kernel<<<(total + 255) / 256, 256, 0, stream>>>(x_in, conv_w,
                                                              conv_b, x_conv);
  }

  // 3) x_proj: (2048 x 96) = x_conv * W_xp^T
  {
    const int N = XDBL_W, K = D_INNER;           // 6 N-tiles -> 1 block/x
    dim3 grid((N / 16 + 7) / 8, M / 16);
    gemm_wmma_kernel<0><<<grid, blk, 0, stream>>>(x_conv, D_INNER, W_xp, D_INNER,
                                                  x_dbl, XDBL_W, nullptr, nullptr,
                                                  M, N, K);
  }

  // 4) dt_proj + softplus: (2048 x 2048) = x_dbl[:, :64] * W_dt^T + b_dt
  {
    const int N = D_INNER, K = DT_RANK;          // 128 N-tiles
    dim3 grid((N / 16 + 7) / 8, M / 16);
    gemm_wmma_kernel<2><<<grid, blk, 0, stream>>>(x_dbl, XDBL_W, W_dt, DT_RANK,
                                                  dtbuf, D_INNER, nullptr, b_dt,
                                                  M, N, K);
  }

  // 5) selective scan + D-skip + gate
  {
    const int total = BATCH * D_INNER;           // 4096
    scan_kernel<<<(total + 255) / 256, 256, 0, stream>>>(dtbuf, x_conv, x_dbl,
                                                         zbuf, A_log, Dp, ybuf);
  }

  // 6) out_proj: (2048 x 1024) = y * W_out^T -> d_out
  {
    const int N = D_MODEL, K = D_INNER;          // 64 N-tiles
    dim3 grid((N / 16 + 7) / 8, M / 16);
    gemm_wmma_kernel<0><<<grid, blk, 0, stream>>>(ybuf, D_INNER, W_out, D_INNER,
                                                  out, D_MODEL, nullptr, nullptr,
                                                  M, N, K);
  }
}
